// Fuse_40510131536240
// MI455X (gfx1250) — compile-verified, matrix operand order checked by reading, not run on hardware
//
#include <hip/hip_runtime.h>

// ---------------------------------------------------------------------------
// B=32, N=1024, DIM=512, H=8, head=64.  SCALE = 0.125.
//
// Algebraic folding: K-projection folded into the single query
// (wk_eff = Wk_h @ q), bk dropped (softmax shift-invariant), V-projection
// pulled outside the attention average (out = (attn@src)@Wv + bv).
// ~1 GFLOP compute, ~262 MB traffic -> bandwidth bound (~11 us @ 23.3 TB/s).
// CDNA5 paths: v_wmma_f32_16x16x32_f16 for the dots GEMM, tensor_load_to_lds
// (TDM) for attn tile staging, NT stores for the write-once output stream.
// ---------------------------------------------------------------------------

#define BB   32
#define NN   1024
#define DIMD 512
#define HH   8
#define HD   64
#define SCL  0.125f

typedef __attribute__((ext_vector_type(16))) _Float16     v16h;
typedef __attribute__((ext_vector_type(8)))  float        v8f;
typedef __attribute__((ext_vector_type(8)))  unsigned int v8u;
typedef __attribute__((ext_vector_type(4)))  float        v4f;
typedef __attribute__((ext_vector_type(4)))  unsigned int v4u;
typedef __attribute__((ext_vector_type(8)))  int          v8i_t;
typedef __attribute__((ext_vector_type(4)))  int          v4i_t;

// ---------------------------------------------------------------------------
// Kernel 1: per (attn, batch): q = score@Wq + bq ; wk_eff[D][h] = Wk[D,h*64+:]·q_h
// Packed into the v_wmma_f32_16x16x32_f16 B-operand register layout:
// 16 k-tiles x 32 lanes x 8 dwords (f16 pairs), heads 8..15 zero-padded.
// ---------------------------------------------------------------------------
__global__ void prep_kernel(const float* __restrict__ score1,
                            const float* __restrict__ score2,
                            const float* __restrict__ Wq,  const float* __restrict__ bq,
                            const float* __restrict__ Wk,
                            const float* __restrict__ Wq1, const float* __restrict__ bq1,
                            const float* __restrict__ Wk1,
                            unsigned int* __restrict__ bpack)
{
    const int blk = blockIdx.x;          // 0..63 = (attn<<5) | batch
    const int a   = blk >> 5;
    const int b   = blk & 31;
    const float* score = (a ? score1 : score2) + b * 16;
    const float* Wq_   = a ? Wq1 : Wq;
    const float* bq_   = a ? bq1 : bq;
    const float* Wk_   = a ? Wk1 : Wk;

    __shared__ float q_s[DIMD];          // 2 KB
    __shared__ float wk_s[DIMD][HH];     // 16 KB

    const int t = threadIdx.x;           // 0..511

    float acc = bq_[t];
#pragma unroll
    for (int c = 0; c < 16; ++c) acc += score[c] * Wq_[c * DIMD + t];
    q_s[t] = acc;
    __syncthreads();

    {
        const int D = t;
        const float* wrow = Wk_ + (size_t)D * DIMD;
        for (int h = 0; h < HH; ++h) {
            float s = 0.f;
            const float* w = wrow + h * HD;
            const float* q = q_s + h * HD;
#pragma unroll 8
            for (int d = 0; d < HD; ++d) s += w[d] * q[d];
            wk_s[D][h] = s;
        }
    }
    __syncthreads();

    {
        const int kt    = t >> 5;
        const int lane  = t & 31;
        const int n     = lane & 15;
        const int koffb = (lane >> 4) ? 8 : 0;
        unsigned int* dst = bpack + ((size_t)blk * 16 + kt) * 256 + lane * 8;
#pragma unroll
        for (int v = 0; v < 8; ++v) {
            const int grp = (v < 4) ? 0 : 16;
            const int k0  = kt * 32 + grp + koffb + (v & 3) * 2;
            union { _Float16 h[2]; unsigned int u; } pk;
            pk.h[0] = (n < HH) ? (_Float16)wk_s[k0][n]     : (_Float16)0.0f;
            pk.h[1] = (n < HH) ? (_Float16)wk_s[k0 + 1][n] : (_Float16)0.0f;
            dst[v] = pk.u;
        }
    }
}

// ---------------------------------------------------------------------------
// Kernel 2: dots[b,h,n] = src[b,n,:] . wk_eff[b,:,h]   via v_wmma_f32_16x16x32_f16.
// grid = (64, 8): blk = (attn,b), 128-row strip; 8 waves/block, each wave owns
// a 16x16 (rows x heads) tile, K = 512/32 = 16 unrolled steps.
// ---------------------------------------------------------------------------
__global__ void dots_kernel(const float* __restrict__ x,
                            const float* __restrict__ l,
                            const unsigned int* __restrict__ bpack,
                            float* __restrict__ dots)
{
    const int blk   = blockIdx.x;
    const int chunk = blockIdx.y;
    const int a     = blk >> 5;
    const int b     = blk & 31;
    const float* src = (a ? l : x) + (size_t)b * NN * DIMD;

    const int wave = threadIdx.x >> 5;
    const int lane = threadIdx.x & 31;
    const int rowbase = chunk * 128 + wave * 16;

    const int m     = lane & 15;
    const int koffb = (lane >> 4) ? 8 : 0;
    const float* arow = src + (size_t)(rowbase + m) * DIMD;
    const unsigned int* bp = bpack + (size_t)blk * 4096 + lane * 8;

    v8f c = {};
#pragma unroll
    for (int kt = 0; kt < 16; ++kt) {
        v16h av;
#pragma unroll
        for (int v = 0; v < 8; ++v) {
            const int grp = (v < 4) ? 0 : 16;
            const int k0  = kt * 32 + grp + koffb + (v & 3) * 2;
            av[2 * v]     = (_Float16)arow[k0];
            av[2 * v + 1] = (_Float16)arow[k0 + 1];
        }
        v8u bu;
#pragma unroll
        for (int v = 0; v < 8; ++v) bu[v] = bp[kt * 256 + v];
        const v16h bvh = __builtin_bit_cast(v16h, bu);
        c = __builtin_amdgcn_wmma_f32_16x16x32_f16(
                false, av, false, bvh, (short)0, c, false, false);
    }

    const int h = lane & 15;
    if (h < HH) {
        const int rbase = rowbase + ((lane >> 4) ? 8 : 0);
        float* drow = dots + ((size_t)blk * HH + h) * NN;
#pragma unroll
        for (int v = 0; v < 8; ++v) drow[rbase + v] = c[v];
    }
}

// ---------------------------------------------------------------------------
// Kernel 3: row softmax over n (512 rows x 1024), SCALE applied here.
// ---------------------------------------------------------------------------
__global__ void softmax_kernel(float* __restrict__ dots)
{
    float* d = dots + (size_t)blockIdx.x * NN;
    const int t = threadIdx.x;           // 256
    __shared__ float red[256];

    float v0 = d[t]       * SCL;
    float v1 = d[t + 256] * SCL;
    float v2 = d[t + 512] * SCL;
    float v3 = d[t + 768] * SCL;

    float m = fmaxf(fmaxf(v0, v1), fmaxf(v2, v3));
    red[t] = m; __syncthreads();
    for (int s = 128; s > 0; s >>= 1) {
        if (t < s) red[t] = fmaxf(red[t], red[t + s]);
        __syncthreads();
    }
    m = red[0]; __syncthreads();

    const float e0 = __expf(v0 - m), e1 = __expf(v1 - m);
    const float e2 = __expf(v2 - m), e3 = __expf(v3 - m);
    red[t] = e0 + e1 + e2 + e3; __syncthreads();
    for (int s = 128; s > 0; s >>= 1) {
        if (t < s) red[t] += red[t + s];
        __syncthreads();
    }
    const float inv = 1.f / red[0];
    d[t] = e0 * inv; d[t + 256] = e1 * inv; d[t + 512] = e2 * inv; d[t + 768] = e3 * inv;
}

// ---------------------------------------------------------------------------
// Kernel 4: partial sbar[h,:] over a 256-row n-chunk.  grid = (64, 4).
// attn tile (8 rows x 256 cols of the 1024-stride attn tensor) is staged into
// LDS with a single TDM tensor_load_to_lds issued by wave 0 (D# per ISA §8.3/8.4),
// completion via s_wait_tensorcnt then block barrier.  Disjoint partial outputs
// keep the reduction bitwise-deterministic (no float atomics).
// ---------------------------------------------------------------------------
__global__ void sbar_kernel(const float* __restrict__ x,
                            const float* __restrict__ l,
                            const float* __restrict__ attn,
                            float* __restrict__ sbar_part)
{
    const int blk = blockIdx.x;          // 0..63
    const int cch = blockIdx.y;          // 0..3 : 256-row n-chunk
    const int a   = blk >> 5;
    const int b   = blk & 31;
    const float* src = (a ? l : x) + (size_t)b * NN * DIMD;
    const float* at  = attn + (size_t)blk * HH * NN + (size_t)cch * 256;

    __shared__ float attn_s[HH * 256];   // 8 KB, compacted [h][256]

    if (threadIdx.x < 32) {
        // Tensor DMA descriptor: 2D tile, tile_dim0=256, tile_dim1=8 rows,
        // tensor_dim0_stride=1024 elements, data_size=4B, compacted into LDS.
        const unsigned int lds_off =
            (unsigned int)(unsigned long long)(uintptr_t)(&attn_s[0]); // addr[31:0] == LDS offset
        const unsigned long long ga = (unsigned long long)(uintptr_t)at;

        v4u g0;
        g0[0] = 1u;                                           // count=1 (valid), user mode
        g0[1] = lds_off;                                      // lds_addr
        g0[2] = (unsigned int)(ga & 0xffffffffu);             // global_addr[31:0]
        g0[3] = (unsigned int)((ga >> 32) & 0x01ffffffu)      // global_addr[56:32]
              | (2u << 30);                                   // type = 2 ("image")

        v8i_t g1;
        g1[0] = (int)(2u << 16);                              // data_size = 2 (4 bytes)
        g1[1] = (int)((NN & 0xffffu) << 16);                  // tensor_dim0[15:0]  (=1024)
        g1[2] = (int)(((unsigned)NN >> 16) | (8u << 16));     // tensor_dim0[31:16], tensor_dim1=8
        g1[3] = (int)(256u << 16);                            // tensor_dim1 hi, tile_dim0=256
        g1[4] = (int)(8u);                                    // tile_dim1=8, tile_dim2=0
        g1[5] = (int)((unsigned)NN);                          // tensor_dim0_stride[31:0] = 1024
        g1[6] = 0;                                            // stride0 hi, stride1[15:0]=0
        g1[7] = 0;

        v4i_t g2 = {};                                        // unused (<=2D tensor)
        v4i_t g3 = {};
        v8i_t g4 = {};                                        // unused trailing group (6-arg form)

        __builtin_amdgcn_tensor_load_to_lds(g0, g1, g2, g3, g4, 0);
        __builtin_amdgcn_s_wait_tensorcnt(0);
    }
    __syncthreads();

    // partial sbar[h][d] = sum over this chunk's 256 n of attn[h][n]*src[n][d]
    const int d = threadIdx.x;           // 0..511, src reads coalesced in d
    const float* srow = src + ((size_t)cch * 256) * DIMD + d;
    float acc[HH] = {};
    for (int n = 0; n < 256; ++n) {
        const float sv = srow[(size_t)n * DIMD];
#pragma unroll
        for (int h = 0; h < HH; ++h) acc[h] += attn_s[h * 256 + n] * sv;
    }
    float* dst = sbar_part + (((size_t)blk * 4 + cch) * HH) * DIMD + d;
#pragma unroll
    for (int h = 0; h < HH; ++h) dst[(size_t)h * DIMD] = acc[h];
}

// ---------------------------------------------------------------------------
// Kernel 5: reduce the 4 partials, then out[col] = bv[col] + sbar[h,:]·Wv[:,col].
// ---------------------------------------------------------------------------
__global__ void proj_kernel(const float* __restrict__ sbar_part,
                            const float* __restrict__ Wv,  const float* __restrict__ bv,
                            const float* __restrict__ Wv1, const float* __restrict__ bv1,
                            float* __restrict__ outs)
{
    const int blk = blockIdx.x;          // 0..63
    const int a   = blk >> 5;
    const float* W    = a ? Wv1 : Wv;
    const float* bias = a ? bv1 : bv;

    __shared__ float sbar[HH * DIMD];    // 16 KB

    const int t = threadIdx.x;           // 512
    for (int i = t; i < HH * DIMD; i += 512) {
        const float* p = sbar_part + ((size_t)blk * 4 * HH) * DIMD + i;
        float s = p[0] + p[HH * DIMD] + p[2 * HH * DIMD] + p[3 * HH * DIMD];
        sbar[i] = s;
    }
    __syncthreads();

    const int h   = t >> 6;
    const int col = t;
    float acc = bias[col];
    for (int D = 0; D < DIMD; ++D) acc += sbar[h * DIMD + D] * W[(size_t)D * DIMD + col];
    outs[(size_t)blk * DIMD + col] = acc;
}

// ---------------------------------------------------------------------------
// Kernel 6: broadcast write (the bandwidth floor).  NT stores: the output is
// write-once/never-read, keep x,l resident in the 192MB L2 across replays.
// ---------------------------------------------------------------------------
__global__ void finalize_kernel(const float* __restrict__ x,
                                const float* __restrict__ l,
                                const float* __restrict__ outs,
                                float* __restrict__ out)
{
    const size_t idx = (size_t)blockIdx.x * blockDim.x + threadIdx.x;  // v4f index
    const int    j4  = (int)(idx & 255);
    const size_t rn  = idx >> 8;
    const int    n   = (int)(rn & 1023);
    const int    b   = (int)(rn >> 10);
    const int    j   = j4 * 4;

    const v4f* sv;
    const v4f* ov;
    if (j < DIMD) {
        sv = reinterpret_cast<const v4f*>(x + ((size_t)b * NN + n) * DIMD + j);
        ov = reinterpret_cast<const v4f*>(outs + (size_t)b * DIMD + j);
    } else {
        sv = reinterpret_cast<const v4f*>(l + ((size_t)b * NN + n) * DIMD + (j - DIMD));
        ov = reinterpret_cast<const v4f*>(outs + (size_t)(32 + b) * DIMD + (j - DIMD));
    }
    const v4f r = *sv + *ov;
    __builtin_nontemporal_store(r, reinterpret_cast<v4f*>(out) + idx);
}

// ---------------------------------------------------------------------------
// Launch
// ---------------------------------------------------------------------------
extern "C" void kernel_launch(void* const* d_in, const int* in_sizes, int n_in,
                              void* d_out, int out_size, void* d_ws, size_t ws_size,
                              hipStream_t stream)
{
    const float* x      = (const float*)d_in[0];
    const float* l      = (const float*)d_in[1];
    const float* score1 = (const float*)d_in[2];
    const float* score2 = (const float*)d_in[3];
    const float* Wq  = (const float*)d_in[4];
    const float* bq  = (const float*)d_in[5];
    const float* Wk  = (const float*)d_in[6];
    // d_in[7] = bk : unused (softmax shift invariance)
    const float* Wv  = (const float*)d_in[8];
    const float* bv  = (const float*)d_in[9];
    const float* Wq1 = (const float*)d_in[10];
    const float* bq1 = (const float*)d_in[11];
    const float* Wk1 = (const float*)d_in[12];
    // d_in[13] = bk1 : unused
    const float* Wv1 = (const float*)d_in[14];
    const float* bv1 = (const float*)d_in[15];
    float* out = (float*)d_out;

    // Workspace layout:
    //   [0, 1MB)    : bpack     (64 * 16 * 32 * 8 dwords) WMMA B operands
    //   [1MB, 3MB)  : dots/attn (64 * 8 * 1024 floats)
    //   [3MB, 7MB)  : sbar_part (64 * 4 * 8 * 512 floats)
    //   [7MB, ...)  : outs      (64 * 512 floats)
    char* ws = (char*)d_ws;
    unsigned int* bpack = (unsigned int*)ws;
    float* dots      = (float*)(ws + (1u << 20));
    float* sbar_part = (float*)(ws + (3u << 20));
    float* outs      = (float*)(ws + (7u << 20));

    prep_kernel<<<64, 512, 0, stream>>>(score1, score2, Wq, bq, Wk, Wq1, bq1, Wk1, bpack);
    dots_kernel<<<dim3(64, 8), 256, 0, stream>>>(x, l, bpack, dots);
    softmax_kernel<<<512, 256, 0, stream>>>(dots);
    sbar_kernel<<<dim3(64, 4), 512, 0, stream>>>(x, l, dots, sbar_part);
    proj_kernel<<<64, 512, 0, stream>>>(sbar_part, Wv, bv, Wv1, bv1, outs);
    finalize_kernel<<<32768, 256, 0, stream>>>(x, l, outs, out);
}
